// ImageEmbed_69595650065102
// MI455X (gfx1250) — compile-verified, matrix operand order checked by reading, not run on hardware
//
#include <hip/hip_runtime.h>

// ---------------- problem constants ----------------
#define BSZ 16
#define TT  1024          // H*W tokens
#define CC  512           // channels
#define NH  8
#define HD  64
#define AD  512           // attention dim
#define EPSV 1e-5f

typedef __bf16 bf16;
typedef __bf16 v16bf __attribute__((ext_vector_type(16)));
typedef float  v8f   __attribute__((ext_vector_type(8)));

// ---------------- WMMA helpers ----------------
__device__ __forceinline__ v8f wmma_bf16(v16bf a, v16bf b, v8f c) {
  // D = A(16x32 bf16) * B(32x16 bf16) + C(16x16 f32)
  return __builtin_amdgcn_wmma_f32_16x16x32_bf16(
      /*neg_a=*/false, a, /*neg_b=*/false, b,
      /*c_mod=*/(short)0, c, /*reuse_a=*/false, /*reuse_b=*/false);
}

// Load a 16x32 bf16 A-operand (or 32x16 B-operand, mirrored) per the CDNA5
// VGPR layout: lanes 0-15 -> row=lane, K chunks {0..7} and {16..23};
// lanes 16-31 -> row=lane-16, K chunks {8..15} and {24..31}.
// Each chunk is 8 contiguous bf16 = one 16-byte load.
__device__ __forceinline__ v16bf load_ab(const bf16* base, int row_stride, int lane) {
  int half = lane >> 4;
  int r    = lane & 15;
  const bf16* p = base + (size_t)r * row_stride + half * 8;
  union { v16bf v; uint4 q[2]; } u;
  u.q[0] = *(const uint4*)(p);        // K 0..7   (or 8..15)
  u.q[1] = *(const uint4*)(p + 16);   // K 16..23 (or 24..31)
  return u.v;
}

// Async-stage a contiguous 16 KB panel (16 rows x 512 bf16) from global into
// LDS: each of 256 threads copies 4 x 16B via global_load_async_to_lds_b128
// (ASYNCcnt path), then waits its own async counter. Caller must __syncthreads.
__device__ __forceinline__ void stage_panel_async(const bf16* __restrict__ gsrc,
                                                  bf16* lds_dst, int tid) {
  unsigned lds_base = (unsigned)(size_t)(void*)lds_dst;  // addr[31:0] = LDS offset
  unsigned off = (unsigned)tid * 16u;
  #pragma unroll
  for (int rr = 0; rr < 4; ++rr) {
    unsigned o = off + (unsigned)rr * 4096u;
    unsigned lds_a = lds_base + o;
    asm volatile("global_load_async_to_lds_b128 %0, %1, %2"
                 :: "v"(lds_a), "v"(o), "s"(gsrc)
                 : "memory");
  }
  asm volatile("s_wait_asynccnt 0x0" ::: "memory");
}

// ---------------- kernel 1: GroupNorm stats (1 group) ----------------
__global__ __launch_bounds__(256) void gn_stats(const float* __restrict__ x,
                                                float* __restrict__ stats) {
  int b = blockIdx.x;
  const float4* p = (const float4*)(x + (size_t)b * TT * CC);
  const int n4 = TT * CC / 4;
  float s = 0.f, ss = 0.f;
  for (int i = threadIdx.x; i < n4; i += 256) {
    float4 v = p[i];
    s  += v.x + v.y + v.z + v.w;
    ss += v.x * v.x + v.y * v.y + v.z * v.z + v.w * v.w;
  }
  __shared__ float sh0[256], sh1[256];
  sh0[threadIdx.x] = s; sh1[threadIdx.x] = ss;
  __syncthreads();
  for (int o = 128; o > 0; o >>= 1) {
    if (threadIdx.x < o) {
      sh0[threadIdx.x] += sh0[threadIdx.x + o];
      sh1[threadIdx.x] += sh1[threadIdx.x + o];
    }
    __syncthreads();
  }
  if (threadIdx.x == 0) {
    const float inv_n = 1.0f / (float)(TT * CC);
    float mean = sh0[0] * inv_n;
    float var  = sh1[0] * inv_n - mean * mean;
    stats[2 * b]     = mean;
    stats[2 * b + 1] = rsqrtf(var + EPSV);
  }
}

// ---------------- kernel 2: normalize + cast to bf16 ----------------
__global__ __launch_bounds__(256) void gn_apply(const float* __restrict__ x,
                                                const float* __restrict__ gamma,
                                                const float* __restrict__ beta,
                                                const float* __restrict__ stats,
                                                bf16* __restrict__ xn) {
  size_t i = ((size_t)blockIdx.x * 256 + threadIdx.x) * 4;
  int b = (int)(i / ((size_t)TT * CC));
  int c = (int)(i % CC);
  float mean = stats[2 * b], inv = stats[2 * b + 1];
  float4 v  = *(const float4*)(x + i);
  float4 g  = *(const float4*)(gamma + c);
  float4 bt = *(const float4*)(beta + c);
  union { bf16 o[4]; uint2 u; } r;
  r.o[0] = (bf16)((v.x - mean) * inv * g.x + bt.x);
  r.o[1] = (bf16)((v.y - mean) * inv * g.y + bt.y);
  r.o[2] = (bf16)((v.z - mean) * inv * g.z + bt.z);
  r.o[3] = (bf16)((v.w - mean) * inv * g.w + bt.w);
  *(uint2*)(xn + i) = r.u;
}

// ---------------- kernel 3: weight transpose + cast (512x512 each) ----------
__global__ __launch_bounds__(256) void wcast_t(const float* __restrict__ wq,
                                               const float* __restrict__ wk,
                                               const float* __restrict__ wv,
                                               const float* __restrict__ wo,
                                               bf16* __restrict__ wqT,
                                               bf16* __restrict__ wkT,
                                               bf16* __restrict__ wvT,
                                               bf16* __restrict__ woT) {
  const float* w = (blockIdx.z == 0) ? wq : (blockIdx.z == 1) ? wk
                   : (blockIdx.z == 2) ? wv : wo;
  bf16* wt = (blockIdx.z == 0) ? wqT : (blockIdx.z == 1) ? wkT
             : (blockIdx.z == 2) ? wvT : woT;
  __shared__ float tile[32][33];
  int bx = blockIdx.x * 32, by = blockIdx.y * 32;
  int tx = threadIdx.x, ty = threadIdx.y;     // 32 x 8
  #pragma unroll
  for (int j = 0; j < 32; j += 8)
    tile[ty + j][tx] = w[(size_t)(by + ty + j) * 512 + bx + tx];
  __syncthreads();
  #pragma unroll
  for (int j = 0; j < 32; j += 8)   // wt[n][k] = w[k][n]
    wt[(size_t)(bx + ty + j) * 512 + by + tx] = (bf16)tile[tx][ty + j];
}

// ---------------- kernel 4: fused QKV WMMA GEMM ----------------
// A panel (16 tokens x 512 K) async-staged in LDS once per block; 8 waves x
// 3 passes x 4 register-blocked accumulator tiles (one shared A per K-step).
// q,k stored [B*T, AD] bf16 ; v stored transposed [B, AD, T] bf16
__global__ __launch_bounds__(256) void qkv_gemm(const bf16* __restrict__ xn,
                                                const bf16* __restrict__ wqT,
                                                const bf16* __restrict__ wkT,
                                                const bf16* __restrict__ wvT,
                                                const float* __restrict__ bq,
                                                const float* __restrict__ bk,
                                                const float* __restrict__ bv,
                                                bf16* __restrict__ q,
                                                bf16* __restrict__ k,
                                                bf16* __restrict__ vT) {
  int t0   = blockIdx.x * 16;            // global token row (over B*T)
  int b    = t0 / TT;
  int tloc = t0 - b * TT;
  int tid  = threadIdx.x;
  int wave = tid >> 5, lane = tid & 31;

  __shared__ __align__(16) bf16 As[16 * CC];   // 16 KB A panel
  stage_panel_async(xn + (size_t)t0 * CC, As, tid);
  __syncthreads();

  const int n = lane & 15, half = lane >> 4;

  for (int pass = 0; pass < 3; ++pass) {
    int tbase = wave * 12 + pass * 4;    // 4 consecutive tiles of 96
    const bf16* wp[4];
    const float* bp[4];
    int nb[4], mt[4];
    #pragma unroll
    for (int t = 0; t < 4; ++t) {
      int tile = tbase + t;
      mt[t] = tile >> 5;                 // 0=q 1=k 2=v
      nb[t] = (tile & 31) * 16;
      const bf16* wt = (mt[t] == 0) ? wqT : (mt[t] == 1) ? wkT : wvT;
      wp[t] = wt + (size_t)nb[t] * CC;
      bp[t] = (mt[t] == 0) ? bq : (mt[t] == 1) ? bk : bv;
    }
    v8f acc[4] = {{}, {}, {}, {}};
    for (int kk = 0; kk < CC; kk += 32) {
      v16bf a = load_ab(As + kk, CC, lane);     // LDS (ds_load_b128)
      #pragma unroll
      for (int t = 0; t < 4; ++t) {
        v16bf bm = load_ab(wp[t] + kk, CC, lane);
        acc[t] = wmma_bf16(a, bm, acc[t]);
      }
    }
    #pragma unroll
    for (int t = 0; t < 4; ++t) {
      float bias = bp[t][nb[t] + n];
      if (mt[t] < 2) {
        bf16* dst = (mt[t] == 0) ? q : k;
        #pragma unroll
        for (int r = 0; r < 8; r++) {
          int m = r + half * 8;          // C-layout row
          dst[(size_t)(t0 + m) * AD + nb[t] + n] = (bf16)(acc[t][r] + bias);
        }
      } else {                           // V transposed: [b][attn_dim][token]
        #pragma unroll
        for (int r = 0; r < 8; r++) {
          int m = r + half * 8;
          vT[((size_t)b * AD + nb[t] + n) * TT + tloc + m] = (bf16)(acc[t][r] + bias);
        }
      }
    }
  }
}

// ---------------- kernel 5: flash attention, 1 wave per (b,h,16-q tile) ----
__global__ __launch_bounds__(32) void attn_flash(const bf16* __restrict__ q,
                                                 const bf16* __restrict__ k,
                                                 const bf16* __restrict__ vT,
                                                 bf16* __restrict__ attn) {
  const int qt = blockIdx.x * 16;   // query tile within batch
  const int h  = blockIdx.y;
  const int b  = blockIdx.z;
  const int lane = threadIdx.x;
  const float scale = 0.125f;       // 1/sqrt(64)

  const bf16* qbase = q  + ((size_t)(b * TT + qt) * AD) + h * HD;
  const bf16* kbase = k  + ((size_t)b * TT) * AD + h * HD;
  const bf16* vbase = vT + ((size_t)b * AD + h * HD) * TT;  // rows = head dim, stride T

  // Q A-operands for K chunks 0..31 and 32..63 (loaded once)
  v16bf a0 = load_ab(qbase,      AD, lane);
  v16bf a1 = load_ab(qbase + 32, AD, lane);

  v8f o[4] = {{}, {}, {}, {}};      // 16x64 output accumulator (4 N-tiles)
  float mrow[8], lrow[8];
  #pragma unroll
  for (int r = 0; r < 8; r++) { mrow[r] = -1e30f; lrow[r] = 0.f; }

  __shared__ __align__(16) bf16 pbuf[16 * 32];   // P tile staging (layout swap)
  const int n = lane & 15, half = lane >> 4;

  for (int kt = 0; kt < TT; kt += 32) {
    // ---- S = Q * K^T for 32 keys (two 16-key N-tiles) ----
    v8f s0 = {}, s1 = {};
    {
      v16bf b0a = load_ab(kbase + (size_t)kt * AD,             AD, lane);
      v16bf b0b = load_ab(kbase + (size_t)kt * AD + 32,        AD, lane);
      v16bf b1a = load_ab(kbase + (size_t)(kt + 16) * AD,      AD, lane);
      v16bf b1b = load_ab(kbase + (size_t)(kt + 16) * AD + 32, AD, lane);
      s0 = wmma_bf16(a0, b0a, s0); s0 = wmma_bf16(a1, b0b, s0);
      s1 = wmma_bf16(a0, b1a, s1); s1 = wmma_bf16(a1, b1b, s1);
    }
    // ---- online softmax (rows live in: vgpr r, 16-lane half) ----
    float mx[8];
    #pragma unroll
    for (int r = 0; r < 8; r++) {
      s0[r] *= scale; s1[r] *= scale;
      mx[r] = fmaxf(s0[r], s1[r]);
    }
    #pragma unroll
    for (int r = 0; r < 8; r++) {
      #pragma unroll
      for (int off = 1; off < 16; off <<= 1)          // stays within half
        mx[r] = fmaxf(mx[r], __shfl_xor(mx[r], off, 32));
      float mnew = fmaxf(mrow[r], mx[r]);
      float corr = __expf(mrow[r] - mnew);
      float p0 = __expf(s0[r] - mnew);
      float p1 = __expf(s1[r] - mnew);
      s0[r] = p0; s1[r] = p1;
      float rs = p0 + p1;
      #pragma unroll
      for (int off = 1; off < 16; off <<= 1)
        rs += __shfl_xor(rs, off, 32);
      lrow[r] = lrow[r] * corr + rs;
      mrow[r] = mnew;
      o[0][r] *= corr; o[1][r] *= corr; o[2][r] *= corr; o[3][r] *= corr;
    }
    // ---- C-layout P -> LDS -> A-layout bf16 ----
    #pragma unroll
    for (int r = 0; r < 8; r++) {
      int m = r + half * 8;
      pbuf[m * 32 + n]      = (bf16)s0[r];
      pbuf[m * 32 + 16 + n] = (bf16)s1[r];
    }
    __syncthreads();
    v16bf pa = load_ab(pbuf, 32, lane);
    // ---- O += P(16x32) * V(32x64), 4 N-tiles of head dim ----
    #pragma unroll
    for (int dt = 0; dt < 4; dt++) {
      v16bf vv = load_ab(vbase + (size_t)(dt * 16) * TT + kt, TT, lane);
      o[dt] = wmma_bf16(pa, vv, o[dt]);
    }
    __syncthreads();
  }

  // ---- normalize and store attn [B*T, AD] bf16 ----
  bf16* outb = attn + ((size_t)(b * TT + qt) * AD) + h * HD;
  #pragma unroll
  for (int r = 0; r < 8; r++) {
    int m = r + half * 8;
    float inv = 1.0f / lrow[r];
    #pragma unroll
    for (int dt = 0; dt < 4; dt++)
      outb[(size_t)m * AD + dt * 16 + n] = (bf16)(o[dt][r] * inv);
  }
}

// ---------------- kernel 6: output projection + bias + residual ------------
// attn panel async-staged in LDS; each wave owns 4 register-blocked N-tiles.
__global__ __launch_bounds__(256) void out_proj(const bf16* __restrict__ attn,
                                                const bf16* __restrict__ woT,
                                                const float* __restrict__ bo,
                                                const float* __restrict__ x,
                                                float* __restrict__ out) {
  int t0  = blockIdx.x * 16;         // token row over B*T
  int tid = threadIdx.x;
  int wave = tid >> 5, lane = tid & 31;

  __shared__ __align__(16) bf16 As[16 * AD];   // 16 KB attn panel
  stage_panel_async(attn + (size_t)t0 * AD, As, tid);
  __syncthreads();

  int nbase0 = wave * 64;            // wave covers n-tiles 4w..4w+3
  v8f acc[4] = {{}, {}, {}, {}};
  for (int kk = 0; kk < AD; kk += 32) {
    v16bf a = load_ab(As + kk, AD, lane);      // LDS
    #pragma unroll
    for (int t = 0; t < 4; ++t) {
      v16bf bm = load_ab(woT + (size_t)(nbase0 + t * 16) * AD + kk, AD, lane);
      acc[t] = wmma_bf16(a, bm, acc[t]);
    }
  }
  int n = lane & 15, half = lane >> 4;
  #pragma unroll
  for (int t = 0; t < 4; ++t) {
    int nbase = nbase0 + t * 16;
    float bias = bo[nbase + n];
    #pragma unroll
    for (int r = 0; r < 8; r++) {
      int m = r + half * 8;
      size_t idx = (size_t)(t0 + m) * CC + nbase + n;
      out[idx] = x[idx] + acc[t][r] + bias;
    }
  }
}

// ---------------- launch ----------------
extern "C" void kernel_launch(void* const* d_in, const int* in_sizes, int n_in,
                              void* d_out, int out_size, void* d_ws, size_t ws_size,
                              hipStream_t stream) {
  const float* x     = (const float*)d_in[0];
  const float* gamma = (const float*)d_in[1];
  const float* beta  = (const float*)d_in[2];
  const float* wq    = (const float*)d_in[3];
  const float* bq    = (const float*)d_in[4];
  const float* wk    = (const float*)d_in[5];
  const float* bk    = (const float*)d_in[6];
  const float* wv    = (const float*)d_in[7];
  const float* bv    = (const float*)d_in[8];
  const float* wo    = (const float*)d_in[9];
  const float* bo    = (const float*)d_in[10];
  float* out = (float*)d_out;

  // workspace carve-up
  const size_t XN = (size_t)BSZ * TT * CC;   // 8.4M elems
  const size_t WW = (size_t)512 * 512;
  char* ws = (char*)d_ws;
  float* stats = (float*)ws;                 // 2*B floats
  bf16* xn  = (bf16*)(ws + 256);
  bf16* wqT = xn  + XN;
  bf16* wkT = wqT + WW;
  bf16* wvT = wkT + WW;
  bf16* woT = wvT + WW;
  bf16* qb  = woT + WW;
  bf16* kb  = qb  + XN;
  bf16* vT  = kb  + XN;
  bf16* attn = xn;                           // reuse xn after QKV

  gn_stats<<<BSZ, 256, 0, stream>>>(x, stats);
  gn_apply<<<(unsigned)(XN / 4 / 256), 256, 0, stream>>>(x, gamma, beta, stats, xn);
  wcast_t<<<dim3(16, 16, 4), dim3(32, 8), 0, stream>>>(wq, wk, wv, wo,
                                                       wqT, wkT, wvT, woT);
  qkv_gemm<<<(BSZ * TT) / 16, 256, 0, stream>>>(xn, wqT, wkT, wvT,
                                                bq, bk, bv, qb, kb, vT);
  attn_flash<<<dim3(TT / 16, NH, BSZ), 32, 0, stream>>>(qb, kb, vT, attn);
  out_proj<<<(BSZ * TT) / 16, 256, 0, stream>>>(attn, woT, bo, x, out);
}
